// CKANConv2d_62929860821097
// MI455X (gfx1250) — compile-verified, matrix operand order checked by reading, not run on hardware
//
#include <hip/hip_runtime.h>
#include <hip/hip_bf16.h>

// ---------------- problem constants ----------------
#define CIN   64
#define COUT  128
#define H_IN  56
#define W_IN  56
#define HO    54
#define WO    54
#define PIX   (HO*WO)          // 2916
#define BATCH 8
#define NROWS (BATCH*PIX)      // 23328 = 32 * 729
#define INF   576              // IN_FEATURES = CIN*3*3
#define NBAS  8
#define KT_BASE 18             // 576/32
#define KT_SPL  144            // 4608/32
#define R1 (KT_BASE/2)         // 9  rounds, 64-K each
#define R2 (KT_SPL/2)          // 72 rounds, 64-K each
#define MTILE 32
#define NBLOCKS (NROWS/MTILE)  // 729 exact
#define PSTRIDE 584            // padded row stride (ushorts), 16B aligned, low bank conflict

typedef __attribute__((ext_vector_type(16))) __bf16        v16bf;
typedef __attribute__((ext_vector_type(8)))  float         v8f;
typedef __attribute__((ext_vector_type(4)))  unsigned int  v4u;

__device__ __forceinline__ unsigned short f2bf(float f) {   // RNE, for weights/p-cache
  unsigned int u = __builtin_bit_cast(unsigned int, f);
  u += 0x7FFFu + ((u >> 16) & 1u);
  return (unsigned short)(u >> 16);
}
__device__ __forceinline__ float bf2f(unsigned short u) {
  unsigned int v = ((unsigned int)u) << 16;
  return __builtin_bit_cast(float, v);
}
// pack two floats into {hi: bf16(a), lo: bf16(b)} with one v_perm_b32 (truncate)
__device__ __forceinline__ unsigned int pk_bf(float a, float b) {
  return __builtin_amdgcn_perm(__builtin_bit_cast(unsigned int, a),
                               __builtin_bit_cast(unsigned int, b), 0x07060302u);
}
// WMMA 16-bit operand element map (ISA 7.12.2): lane L, elem e -> K
__device__ __forceinline__ int kmap(int lane, int e) {
  return (e < 8 ? e : e + 8) + (lane >= 16 ? 8 : 0);
}
__device__ __forceinline__ float fast_silu(float v) {
  float e = __expf(-v);
  return v * __builtin_amdgcn_rcpf(1.0f + e);
}

// Closed-form cubic B-spline: knots g[t] = -2.2 + 0.4 t.  For v in interval t
// only bases t-3..t are non-zero with the standard uniform cubic weights.
// Emits the 8 bases as 4 bf16-packed dwords (dword d = {bas[2d+1], bas[2d]}).
__device__ __forceinline__ void bspline8_pk(float v, unsigned int pk[4]) {
  float s  = (v + 2.2f) * 2.5f;       // knot units
  float fs = floorf(s);
  int   t  = (int)fs;                 // interval; out-of-range t selects all-zero below
  float u  = s - fs;
  float um = 1.0f - u;
  float u2 = u * u, u3 = u2 * u;
  float w3 = u3 * (1.0f / 6.0f);
  float w0 = um * um * um * (1.0f / 6.0f);
  float w1 = (0.5f * u3 - u2) + (2.0f / 3.0f);        // (3u^3-6u^2+4)/6
  float w2 = ((1.0f - w0) - w1) - w3;                 // partition of unity
  unsigned int E0 = pk_bf(w1, w0);   // {w1,w0}
  unsigned int E1 = pk_bf(w3, w2);   // {w3,w2}
  unsigned int O0 = pk_bf(w2, w1);   // {w2,w1}
  unsigned int On = E0 << 16;        // {w0, 0}
  unsigned int O1 = E1 >> 16;        // {0, w3}
#pragma unroll
  for (int d = 0; d < 4; ++d) {
    unsigned int r = 0u;
    r = (t == 2 * d + 4) ? On : r;   // bas[2d]=0,  bas[2d+1]=w0
    r = (t == 2 * d + 3) ? E0 : r;   // w0, w1
    r = (t == 2 * d + 2) ? O0 : r;   // w1, w2
    r = (t == 2 * d + 1) ? E1 : r;   // w2, w3
    r = (t == 2 * d    ) ? O1 : r;   // w3, 0
    pk[d] = r;
  }
}

// ---------------------------------------------------------------
// Kernel 1: combined bf16 weights, pre-swizzled into WMMA B-fragment
// order [ktile][otile(8)][lane(32)][e(16)]; base block then spline block.
// ---------------------------------------------------------------
__global__ void prep_weights(const float* __restrict__ bw,
                             const float* __restrict__ sw,
                             const float* __restrict__ sc,
                             unsigned short* __restrict__ wsw) {
  const int NBASE = KT_BASE * 8 * 32 * 16;   // 73728
  const int NSPL  = KT_SPL  * 8 * 32 * 16;   // 589824
  int t = blockIdx.x * 256 + threadIdx.x;
  if (t < NBASE) {
    int e = t & 15, lane = (t >> 4) & 31, ot = (t >> 9) & 7, kt = t >> 12;
    int k = kt * 32 + kmap(lane, e);
    int o = ot * 16 + (lane & 15);
    wsw[t] = f2bf(bw[o * INF + k]);
  } else if (t < NBASE + NSPL) {
    int t2 = t - NBASE;
    int e = t2 & 15, lane = (t2 >> 4) & 31, ot = (t2 >> 9) & 7, kt = t2 >> 12;
    int K2 = kt * 32 + kmap(lane, e);
    int i = K2 >> 3, g = K2 & 7;
    int o = ot * 16 + (lane & 15);
    wsw[t] = f2bf(sw[(o * INF + i) * NBAS + g] * sc[o * INF + i]);
  }
}

// ---------------------------------------------------------------
// Kernel 2: fused activation + bf16 WMMA GEMM.
// Block = 256 thr (8 waves): 32 rows x 128 out channels.
// Each round covers 64 K-values (two 32-K chunks), 4 WMMAs per wave.
// ---------------------------------------------------------------
__global__ void __launch_bounds__(256)
kan_gemm(const float* __restrict__ x,
         const unsigned short* __restrict__ wsw,
         float* __restrict__ out) {
  __shared__ __align__(32) unsigned short p_lds[MTILE * PSTRIDE]; // bf16 p, [row][i], 36.5KB
  __shared__ __align__(32) unsigned short a_lds[2][1024];         // two A-fragment chunks, 4KB

  const int tid  = threadIdx.x;
  const int lane = tid & 31;
  const int wv   = tid >> 5;
  const int wm   = wv & 1;         // m-subtile (16 rows)
  const int wn   = wv >> 1;        // 0..3 -> cols [wn*32, wn*32+32)
  const int r0   = blockIdx.x * MTILE;

  // ---- fill patch cache (unfold): one row per thread-slot, coalesced in lr ----
  {
    const int lr  = tid & 31;
    const int i0  = tid >> 5;          // 0..7
    const int row = r0 + lr;
    const int b   = row / PIX;
    const int pix = row - b * PIX;
    const int ho = pix / WO, wo = pix - ho * WO;
    const float* xrow = x + ((b * CIN) * H_IN + ho) * W_IN + wo;  // + c*3136 + di*56 + dj
    unsigned short* prow = p_lds + lr * PSTRIDE;
    for (int s = 0; s < INF / 8; ++s) {
      int i  = i0 + s * 8;
      int c  = i / 9;
      int kk = i - c * 9;
      int di = kk / 3, dj = kk - di * 3;
      prow[i] = f2bf(xrow[c * (H_IN * W_IN) + di * W_IN + dj]);
    }
  }

  v8f acc0 = {};
  v8f acc1 = {};

  const int q  = tid >> 5;               // staging octet / feature id (0..7)
  const int lr = tid & 31;               // staging row
  const int hh = q >> 2;                 // chunk within round
  const int ql = q & 3;
  const int la = (lr & 15) + ((ql & 1) << 4);
  const int e0 = (ql >> 1) << 3;
  unsigned short* a_dst = &a_lds[hh][((lr >> 4) * 32 + la) * 16 + e0];
  const int afrag_off = (wm * 32 + lane) * 16;
  const int bfrag_off = ((wn * 2) * 32 + lane) * 16;   // + kt*8*32*16

  // ================= phase 1: silu(p) @ base_weight^T (9 rounds) =================
  for (int r = 0; r < R1; ++r) {
    // B fragments for this round: issue early, overlap with staging VALU
    const unsigned short* bp0 = wsw + (2 * r + 0) * 4096 + bfrag_off;
    const unsigned short* bp1 = wsw + (2 * r + 1) * 4096 + bfrag_off;
    v16bf b00 = *(const v16bf*)bp0;
    v16bf b01 = *(const v16bf*)(bp0 + 512);
    v16bf b10 = *(const v16bf*)bp1;
    v16bf b11 = *(const v16bf*)(bp1 + 512);
    __syncthreads();
    {   // stage 32 rows x 64 K: thread handles 8 consecutive K for one row
      v4u pv = *(const v4u*)&p_lds[lr * PSTRIDE + r * 64 + q * 8];
      v4u od;
#pragma unroll
      for (int j = 0; j < 4; ++j) {
        unsigned int d = pv[j];
        float flo = __builtin_bit_cast(float, d << 16);
        float fhi = __builtin_bit_cast(float, d & 0xFFFF0000u);
        od[j] = pk_bf(fast_silu(fhi), fast_silu(flo));
      }
      *(v4u*)a_dst = od;
    }
    __syncthreads();
    v16bf af0 = *(const v16bf*)&a_lds[0][afrag_off];
    v16bf af1 = *(const v16bf*)&a_lds[1][afrag_off];
    acc0 = __builtin_amdgcn_wmma_f32_16x16x32_bf16(false, af0, false, b00, (short)0, acc0, false, false);
    acc1 = __builtin_amdgcn_wmma_f32_16x16x32_bf16(false, af0, false, b01, (short)0, acc1, false, false);
    acc0 = __builtin_amdgcn_wmma_f32_16x16x32_bf16(false, af1, false, b10, (short)0, acc0, false, false);
    acc1 = __builtin_amdgcn_wmma_f32_16x16x32_bf16(false, af1, false, b11, (short)0, acc1, false, false);
  }

  // ============ phase 2: bases(p) @ (spline_weight*scaler)^T (72 rounds) ============
  const unsigned short* wspl = wsw + KT_BASE * 4096;
  for (int r = 0; r < R2; ++r) {
    const unsigned short* bp0 = wspl + (2 * r + 0) * 4096 + bfrag_off;
    const unsigned short* bp1 = wspl + (2 * r + 1) * 4096 + bfrag_off;
    v16bf b00 = *(const v16bf*)bp0;
    v16bf b01 = *(const v16bf*)(bp0 + 512);
    v16bf b10 = *(const v16bf*)bp1;
    v16bf b11 = *(const v16bf*)(bp1 + 512);
    __syncthreads();
    {   // stage: thread = one (row, feature); 8 bases -> one b128 store
      float v = bf2f(p_lds[lr * PSTRIDE + r * 8 + q]);
      unsigned int pk[4];
      bspline8_pk(v, pk);
      v4u od = {pk[0], pk[1], pk[2], pk[3]};
      *(v4u*)a_dst = od;
    }
    __syncthreads();
    v16bf af0 = *(const v16bf*)&a_lds[0][afrag_off];
    v16bf af1 = *(const v16bf*)&a_lds[1][afrag_off];
    acc0 = __builtin_amdgcn_wmma_f32_16x16x32_bf16(false, af0, false, b00, (short)0, acc0, false, false);
    acc1 = __builtin_amdgcn_wmma_f32_16x16x32_bf16(false, af0, false, b01, (short)0, acc1, false, false);
    acc0 = __builtin_amdgcn_wmma_f32_16x16x32_bf16(false, af1, false, b10, (short)0, acc0, false, false);
    acc1 = __builtin_amdgcn_wmma_f32_16x16x32_bf16(false, af1, false, b11, (short)0, acc1, false, false);
  }

  // ---- store C (VGPR r -> M = r + 8*(lane>=16), N = lane%16) ----
  const int col0 = wn * 32 + (lane & 15);
#pragma unroll
  for (int r = 0; r < 8; ++r) {
    int m   = r + ((lane >> 4) << 3);
    int row = r0 + wm * 16 + m;
    int b   = row / PIX;
    int pix = row - b * PIX;
    out[(b * COUT + col0) * PIX + pix]      = acc0[r];
    out[(b * COUT + col0 + 16) * PIX + pix] = acc1[r];
  }
}

extern "C" void kernel_launch(void* const* d_in, const int* in_sizes, int n_in,
                              void* d_out, int out_size, void* d_ws, size_t ws_size,
                              hipStream_t stream) {
  const float* x  = (const float*)d_in[0];   // (8,64,56,56)
  const float* bw = (const float*)d_in[1];   // (128,576)
  const float* sw = (const float*)d_in[2];   // (128,576,8)
  const float* sc = (const float*)d_in[3];   // (128,576)
  float* out = (float*)d_out;                // (8,128,54,54)
  unsigned short* wsw = (unsigned short*)d_ws; // 1.33 MB swizzled bf16 weights

  const int total = (KT_BASE + KT_SPL) * 8 * 32 * 16;  // 663552
  prep_weights<<<(total + 255) / 256, 256, 0, stream>>>(bw, sw, sc, wsw);
  kan_gemm<<<NBLOCKS, 256, 0, stream>>>(x, wsw, out);
}
